// Half_Graph_51539607552497
// MI455X (gfx1250) — compile-verified, matrix operand order checked by reading, not run on hardware
//
#include <hip/hip_runtime.h>
#include <hip/hip_bf16.h>

typedef __attribute__((ext_vector_type(16))) _Float16 v16h;
typedef __attribute__((ext_vector_type(8)))  _Float16 v8h;
typedef __attribute__((ext_vector_type(8)))  float    v8f;

#define NTOT 18432          // B*H*W
#define HW   9216
#define IMG_W 96
#define IMG_H 96
#define EPSV 1e-5f

// fast sigmoid: v_exp_f32 + v_rcp_f32 (avoids IEEE div Newton iteration)
__device__ __forceinline__ float sigmoidf_(float v) {
    return __builtin_amdgcn_rcpf(1.0f + __expf(-v));
}

// ---------------------------------------------------------------------------
// small utility kernels
// ---------------------------------------------------------------------------
__global__ void zero_f16(_Float16* p, int n) {
    int i = blockIdx.x * blockDim.x + threadIdx.x;
    if (i < n) p[i] = (_Float16)0.0f;
}

// pack (O, CIN, 3, 3) f32 weights -> f16 [32][9*CINP] rows=M, col = tap*CINP + c
__global__ void pack_w3x3(const float* __restrict__ w, _Float16* __restrict__ dst,
                          int O, int CIN, int CINP, int total) {
    int idx = blockIdx.x * blockDim.x + threadIdx.x;
    if (idx >= total) return;
    int m = idx / (9 * CINP);
    int r = idx % (9 * CINP);
    int t = r / CINP, c = r % CINP;
    float v = (m < O && c < CIN) ? w[((size_t)m * CIN + c) * 9 + t] : 0.0f;
    dst[idx] = (_Float16)v;
}

// T[n][c] f16, c in [0,32): cat(A,B) 10+10 channels, pad zero
__global__ void build_T_cat2(const float* __restrict__ A, const float* __restrict__ Bm,
                             _Float16* __restrict__ T) {
    int idx = blockIdx.x * blockDim.x + threadIdx.x;
    if (idx >= NTOT * 32) return;
    int n = idx >> 5, c = idx & 31;
    int b = n / HW, p = n % HW;
    float v = 0.0f;
    if (c < 10)      v = A [((size_t)b * 10 + c) * HW + p];
    else if (c < 20) v = Bm[((size_t)b * 10 + (c - 10)) * HW + p];
    T[idx] = (_Float16)v;
}

// T[n][c] f16, c in [0,320): cat(fea256, x10, comp10, dec10, dp10), pad zero
__global__ void build_T_cat5(const float* __restrict__ fea, const float* __restrict__ x,
                             const float* __restrict__ comp, const float* __restrict__ dec,
                             const float* __restrict__ dp, _Float16* __restrict__ T) {
    int idx = blockIdx.x * blockDim.x + threadIdx.x;
    if (idx >= NTOT * 320) return;
    int n = idx / 320, c = idx % 320;
    int b = n / HW, p = n % HW;
    float v = 0.0f;
    if (c < 256)      v = fea[((size_t)b * 256 + c) * HW + p];
    else if (c < 266) v = x [((size_t)b * 10 + (c - 256)) * HW + p];
    else if (c < 276) v = comp[(size_t)(c - 266) * NTOT + n];
    else if (c < 286) v = dec [(size_t)(c - 276) * NTOT + n];
    else if (c < 296) v = dp  [(size_t)(c - 286) * NTOT + n];
    T[idx] = (_Float16)v;
}

// ---------------------------------------------------------------------------
// pointwise fused kernels (1x1 conv paths)
// ---------------------------------------------------------------------------
__global__ void composition_kernel(const float* __restrict__ xp, int part0, int nparts,
                                   const float* __restrict__ w,
                                   const float* __restrict__ g, const float* __restrict__ bb,
                                   const float* __restrict__ bm, const float* __restrict__ bv,
                                   float* __restrict__ out) {
    int n = blockIdx.x * blockDim.x + threadIdx.x;
    if (n >= NTOT) return;
    int b = n / HW, p = n % HW;
    float acc[10];
#pragma unroll
    for (int o = 0; o < 10; ++o) acc[o] = 0.0f;
    int K = nparts * 10;
    for (int i = 0; i < nparts; ++i) {
#pragma unroll
        for (int c = 0; c < 10; ++c) {
            float v = xp[((size_t)(b * 6 + part0 + i) * 10 + c) * HW + p];
            int k = i * 10 + c;
#pragma unroll
            for (int o = 0; o < 10; ++o) acc[o] += w[o * K + k] * v;
        }
    }
#pragma unroll
    for (int o = 0; o < 10; ++o) {
        float t = g[o] * (acc[o] - bm[o]) * rsqrtf(bv[o] + EPSV) + bb[o];
        out[(size_t)o * NTOT + n] = fmaxf(t, 0.0f);
    }
}

__global__ void decomposition_kernel(const float* __restrict__ xf, const float* __restrict__ xh,
                                     const float* __restrict__ w1,
                                     const float* __restrict__ g, const float* __restrict__ bb,
                                     const float* __restrict__ bm, const float* __restrict__ bv,
                                     const float* __restrict__ w2, const float* __restrict__ b2,
                                     float* __restrict__ dec, float* __restrict__ att_out, int hsel) {
    int n = blockIdx.x * blockDim.x + threadIdx.x;
    if (n >= NTOT) return;
    int b = n / HW, p = n % HW;
    float xfv[10];
#pragma unroll
    for (int c = 0; c < 10; ++c) xfv[c] = xf[((size_t)b * 10 + c) * HW + p];
    float a = b2[0];
#pragma unroll
    for (int j = 0; j < 10; ++j) {
        float s = 0.0f;
#pragma unroll
        for (int c = 0; c < 10; ++c) s += w1[j * 20 + c] * xfv[c];
#pragma unroll
        for (int c = 0; c < 10; ++c) s += w1[j * 20 + 10 + c] * xh[((size_t)b * 10 + c) * HW + p];
        float hj = fmaxf(g[j] * (s - bm[j]) * rsqrtf(bv[j] + EPSV) + bb[j], 0.0f);
        a += w2[j] * hj;
    }
    a = sigmoidf_(a);
    att_out[(size_t)(b * 2 + hsel) * HW + p] = a;
#pragma unroll
    for (int c = 0; c < 10; ++c) dec[(size_t)c * NTOT + n] = a * xfv[c];
}

__global__ void dp_finalize_kernel(const float* __restrict__ pA,
                                   const float* __restrict__ aw, const float* __restrict__ ab,
                                   const float* __restrict__ d2h, float* __restrict__ dpmap) {
    int n = blockIdx.x * blockDim.x + threadIdx.x;
    if (n >= NTOT) return;
    int b = n / HW, p = n % HW;
    float a = ab[0];
#pragma unroll
    for (int c = 0; c < 10; ++c) a += aw[c] * pA[((size_t)b * 10 + c) * HW + p];
    float s = 2.0f - sigmoidf_(a);
#pragma unroll
    for (int c = 0; c < 10; ++c) dpmap[(size_t)c * NTOT + n] = s * d2h[(size_t)c * NTOT + n];
}

__global__ void final_combine_kernel(const float* __restrict__ gate, const float* __restrict__ x,
                                     const float* __restrict__ out2, float* __restrict__ dst) {
    int n = blockIdx.x * blockDim.x + threadIdx.x;
    if (n >= NTOT) return;
    int b = n / HW, p = n % HW;
    float gv = gate[n];
#pragma unroll
    for (int c = 0; c < 10; ++c) {
        float o = gv * x[((size_t)b * 10 + c) * HW + p] + (1.0f - gv) * out2[(size_t)c * NTOT + n];
        dst[((size_t)b * 10 + c) * HW + p] = fmaxf(o, 0.0f);
    }
}

// ---------------------------------------------------------------------------
// WMMA implicit-GEMM 3x3 conv -> 27ch offsets (rows 0-8 off_y, 9-17 off_x,
// 18-26 sigmoid mask), f32 [27][NTOT]
// ---------------------------------------------------------------------------
template<int CINP>
__global__ void __launch_bounds__(256)
conv3x3_offs_wmma(const _Float16* __restrict__ Tin, const _Float16* __restrict__ Wp,
                  const float* __restrict__ off_b, float* __restrict__ offs,
                  const _Float16* __restrict__ zb) {
    int wid  = blockIdx.x * (blockDim.x >> 5) + (threadIdx.x >> 5);
    int lane = threadIdx.x & 31;
    int nbase = wid << 4;
    if (nbase >= NTOT) return;                 // uniform per wave (never hit at 1152 waves)
    int hi = lane >> 4;
    int ncol = nbase + (lane & 15);
    int b = ncol / HW, p = ncol % HW;
    int y = p / IMG_W, x = p % IMG_W;

    v8f acc0 = {}, acc1 = {};
    const int KROW = 9 * CINP;
    const _Float16* arow0 = Wp + (size_t)(lane & 15) * KROW;
    const _Float16* arow1 = arow0 + (size_t)16 * KROW;

#pragma clang loop unroll(disable)
    for (int t = 0; t < 9; ++t) {
        int yy = y + (t / 3) - 1;
        int xx = x + (t % 3) - 1;
        bool valid = (yy >= 0) && (yy < IMG_H) && (xx >= 0) && (xx < IMG_W);
        const _Float16* bbase = Tin + (size_t)(b * HW + yy * IMG_W + xx) * CINP;
        for (int kc = 0; kc < CINP / 32; ++kc) {
            int ko = kc * 32 + hi * 16;
            const _Float16* q = valid ? (bbase + ko) : zb;
            v16h bv = *(const v16h*)q;
            v16h a0 = *(const v16h*)(arow0 + t * CINP + ko);
            acc0 = __builtin_amdgcn_wmma_f32_16x16x32_f16(false, a0, false, bv, (short)0, acc0, false, false);
            v16h a1 = *(const v16h*)(arow1 + t * CINP + ko);
            acc1 = __builtin_amdgcn_wmma_f32_16x16x32_f16(false, a1, false, bv, (short)0, acc1, false, false);
        }
    }
#pragma unroll
    for (int mt = 0; mt < 2; ++mt) {
        v8f acc = mt ? acc1 : acc0;
        int mb = mt * 16 + hi * 8;
#pragma unroll
        for (int j = 0; j < 8; ++j) {
            int m = mb + j;
            if (m < 27) {
                float v = acc[j] + off_b[m];
                if (m >= 18) v = sigmoidf_(v);
                offs[(size_t)m * NTOT + ncol] = v;
            }
        }
    }
}

// ---------------------------------------------------------------------------
// Fused deformable sampling + WMMA einsum.
// B-operand (16 K-values per lane) built in registers from 4 contiguous f16
// corner gathers; the bilinear*mask blend runs on the packed-f16 pipe
// (v_pk_fma_f16) via ext-vector arithmetic with scalar-splat weights.
// EPMODE 0: BN+ReLU -> f16 [N][32]; 1: BN+ReLU -> f32 [COUT][N];
// EPMODE 2: sigmoid(acc + bias) -> f32 [1][N]
// ---------------------------------------------------------------------------
template<int CINP, int NTILES, int EPMODE>
__global__ void __launch_bounds__(256)
dfconv_sample_wmma(const _Float16* __restrict__ Tin, const float* __restrict__ offs,
                   const _Float16* __restrict__ Wp, int COUT,
                   const float* __restrict__ bn_g, const float* __restrict__ bn_b,
                   const float* __restrict__ bn_m, const float* __restrict__ bn_v,
                   const float* __restrict__ bias,
                   _Float16* __restrict__ outT, float* __restrict__ outF,
                   const _Float16* __restrict__ zb) {
    int wid  = blockIdx.x * (blockDim.x >> 5) + (threadIdx.x >> 5);
    int lane = threadIdx.x & 31;
    int nbase = wid << 4;
    if (nbase >= NTOT) return;
    int hi = lane >> 4;
    int ncol = nbase + (lane & 15);
    int b = ncol / HW, p = ncol % HW;
    int y = p / IMG_W, x = p % IMG_W;

    v8f acc0 = {}, acc1 = {};
    const int KROW = 9 * CINP;
    const _Float16* arow0 = Wp + (size_t)(lane & 15) * KROW;
    const _Float16* arow1 = arow0 + (size_t)16 * KROW;

#pragma clang loop unroll(disable)
    for (int t = 0; t < 9; ++t) {
        float offy = offs[(size_t)t * NTOT + ncol];
        float offx = offs[(size_t)(9 + t) * NTOT + ncol];
        float msk  = offs[(size_t)(18 + t) * NTOT + ncol];
        float ys = (float)y + (float)((t / 3) - 1) + offy;
        float xs = (float)x + (float)((t % 3) - 1) + offx;
        float fy = floorf(ys), fx = floorf(xs);
        int y0 = (int)fy, x0 = (int)fx;
        float dy = ys - fy, dx = xs - fx;
        _Float16 w00 = (_Float16)((1.0f - dy) * (1.0f - dx) * msk);
        _Float16 w01 = (_Float16)((1.0f - dy) * dx * msk);
        _Float16 w10 = (_Float16)(dy * (1.0f - dx) * msk);
        _Float16 w11 = (_Float16)(dy * dx * msk);
        bool y0ok = (y0 >= 0) && (y0 < IMG_H);
        bool y1ok = (y0 + 1 >= 0) && (y0 + 1 < IMG_H);
        bool x0ok = (x0 >= 0) && (x0 < IMG_W);
        bool x1ok = (x0 + 1 >= 0) && (x0 + 1 < IMG_W);
        bool ok00 = y0ok && x0ok, ok01 = y0ok && x1ok;
        bool ok10 = y1ok && x0ok, ok11 = y1ok && x1ok;
        const _Float16* b00 = Tin + (size_t)(b * HW + y0 * IMG_W + x0) * CINP;
        const _Float16* b01 = b00 + CINP;
        const _Float16* b10 = b00 + (size_t)IMG_W * CINP;
        const _Float16* b11 = b10 + CINP;

        for (int kc = 0; kc < CINP / 32; ++kc) {
            int ko = kc * 32 + hi * 16;
            const _Float16* q00 = ok00 ? (b00 + ko) : zb;
            const _Float16* q01 = ok01 ? (b01 + ko) : zb;
            const _Float16* q10 = ok10 ? (b10 + ko) : zb;
            const _Float16* q11 = ok11 ? (b11 + ko) : zb;
            v16h v00 = *(const v16h*)q00;
            v16h v01 = *(const v16h*)q01;
            v16h v10 = *(const v16h*)q10;
            v16h v11 = *(const v16h*)q11;
            // packed-f16 blend: lowers to v_pk_fma_f16 on the VOP3P pipe
            v16h bv = v00 * w00 + v01 * w01 + v10 * w10 + v11 * w11;
            v16h a0 = *(const v16h*)(arow0 + t * CINP + ko);
            acc0 = __builtin_amdgcn_wmma_f32_16x16x32_f16(false, a0, false, bv, (short)0, acc0, false, false);
            if (NTILES == 2) {
                v16h a1 = *(const v16h*)(arow1 + t * CINP + ko);
                acc1 = __builtin_amdgcn_wmma_f32_16x16x32_f16(false, a1, false, bv, (short)0, acc1, false, false);
            }
        }
    }
#pragma unroll
    for (int mt = 0; mt < NTILES; ++mt) {
        v8f acc = mt ? acc1 : acc0;
        int mb = mt * 16 + hi * 8;
        if (EPMODE == 0) {
            v8h st;
#pragma unroll
            for (int j = 0; j < 8; ++j) {
                int m = mb + j;
                float val = 0.0f;
                if (m < COUT) {
                    float t2 = bn_g[m] * (acc[j] - bn_m[m]) * rsqrtf(bn_v[m] + EPSV) + bn_b[m];
                    val = fmaxf(t2, 0.0f);
                }
                st[j] = (_Float16)val;
            }
            *(v8h*)(outT + (size_t)ncol * 32 + mb) = st;
        } else {
#pragma unroll
            for (int j = 0; j < 8; ++j) {
                int m = mb + j;
                if (m < COUT) {
                    float val;
                    if (EPMODE == 1) {
                        float t2 = bn_g[m] * (acc[j] - bn_m[m]) * rsqrtf(bn_v[m] + EPSV) + bn_b[m];
                        val = fmaxf(t2, 0.0f);
                    } else {
                        val = sigmoidf_(acc[j] + bias[0]);
                    }
                    outF[(size_t)m * NTOT + ncol] = val;
                }
            }
        }
    }
}

// ---------------------------------------------------------------------------
// host launcher
// ---------------------------------------------------------------------------
extern "C" void kernel_launch(void* const* d_in, const int* in_sizes, int n_in,
                              void* d_out, int out_size, void* d_ws, size_t ws_size,
                              hipStream_t stream) {
    (void)in_sizes; (void)n_in; (void)out_size; (void)ws_size;
    const float* fea = (const float*)d_in[0];
    const float* xf  = (const float*)d_in[1];
    const float* xh0 = (const float*)d_in[2];
    const float* xh1 = (const float*)d_in[3];
    const float* xp  = (const float*)d_in[4];
    auto F = [&](int i) { return (const float*)d_in[i]; };

    // ---- workspace carve (~22 MB) ----
    char* base = (char*)d_ws; size_t off = 0;
    auto carve = [&](size_t bytes) -> void* {
        void* r = base + off; off += (bytes + 255) & ~(size_t)255; return r;
    };
    _Float16* zb = (_Float16*)carve(512);
    const size_t SMALL = (size_t)32 * 9 * 32;    // elements
    const size_t BIG   = (size_t)32 * 9 * 320;
    _Float16 *P_d1_ow[2], *P_d1_w[2], *P_d2_ow[2], *P_d2_w[2];
    for (int h = 0; h < 2; ++h) {
        P_d1_ow[h] = (_Float16*)carve(SMALL * 2);
        P_d1_w [h] = (_Float16*)carve(SMALL * 2);
        P_d2_ow[h] = (_Float16*)carve(SMALL * 2);
        P_d2_w [h] = (_Float16*)carve(SMALL * 2);
    }
    _Float16* P_c1_ow = (_Float16*)carve(BIG * 2);
    _Float16* P_c1_w  = (_Float16*)carve(BIG * 2);
    _Float16* P_c2_ow = (_Float16*)carve(SMALL * 2);
    _Float16* P_c2_w  = (_Float16*)carve(SMALL * 2);
    _Float16* P_g_ow  = (_Float16*)carve(SMALL * 2);
    _Float16* P_g_w   = (_Float16*)carve(SMALL * 2);
    _Float16* T_dp[2];
    T_dp[0] = (_Float16*)carve((size_t)NTOT * 32 * 2);
    T_dp[1] = (_Float16*)carve((size_t)NTOT * 32 * 2);
    _Float16* T_small = (_Float16*)carve((size_t)NTOT * 32 * 2);
    _Float16* T_cat   = (_Float16*)carve((size_t)NTOT * 320 * 2);
    float* offs   = (float*)carve((size_t)27 * NTOT * 4);
    float* comp_m = (float*)carve((size_t)10 * NTOT * 4);
    float* dec_m  = (float*)carve((size_t)10 * NTOT * 4);
    float* dp_m   = (float*)carve((size_t)10 * NTOT * 4);
    float* d2h_m  = (float*)carve((size_t)10 * NTOT * 4);
    float* out2_m = (float*)carve((size_t)10 * NTOT * 4);
    float* gate_m = (float*)carve((size_t)NTOT * 4);

    zero_f16<<<1, 256, 0, stream>>>(zb, 256);

    auto pack = [&](int wi, _Float16* dst, int O, int CIN, int CINP) {
        int total = 32 * 9 * CINP;
        pack_w3x3<<<(total + 255) / 256, 256, 0, stream>>>(F(wi), dst, O, CIN, CINP, total);
    };
    // params flattened: top-level insertion order, params dict sorted-key recursive.
    // dp_u base 45, dp_l base 29; upd: bn1 61-64, bn2 65-68, c1 69-71, c2 72-74, gate 75-78
    pack(56, P_d1_ow[0], 27, 20, 32);  pack(57, P_d1_w[0], 20, 20, 32);
    pack(59, P_d2_ow[0], 27, 20, 32);  pack(60, P_d2_w[0], 10, 20, 32);
    pack(40, P_d1_ow[1], 27, 20, 32);  pack(41, P_d1_w[1], 20, 20, 32);
    pack(43, P_d2_ow[1], 27, 20, 32);  pack(44, P_d2_w[1], 10, 20, 32);
    pack(70, P_c1_ow, 27, 296, 320);   pack(71, P_c1_w, 20, 296, 320);
    pack(73, P_c2_ow, 27, 20, 32);     pack(74, P_c2_w, 10, 20, 32);
    pack(77, P_g_ow,  27, 20, 32);     pack(78, P_g_w,   1, 20, 32);

    {
        int tot = NTOT * 32;
        build_T_cat2<<<(tot + 255) / 256, 256, 0, stream>>>(xh1, xh0, T_dp[0]); // dp_u: [xh1,xh0]
        build_T_cat2<<<(tot + 255) / 256, 256, 0, stream>>>(xh0, xh1, T_dp[1]); // dp_l: [xh0,xh1]
    }

    dim3 gW(144), bW(256);                    // 1152 full waves * 16 cols = 18432
    dim3 gP((NTOT + 255) / 256), bP(256);
    float* out_att = (float*)d_out + 2 * 184320;

    for (int hsel = 0; hsel < 2; ++hsel) {
        const float* x  = (hsel == 0) ? xh0 : xh1;
        const float* pA = (hsel == 0) ? xh1 : xh0;
        int comp_base = (hsel == 0) ? 10 : 5;       // bn(g,b,m,v), w
        int part0  = (hsel == 0) ? 0 : 4;
        int nparts = (hsel == 0) ? 4 : 2;
        int dec    = (hsel == 0) ? 22 : 15;         // b2, bn1(g,b,m,v), w1, w2
        int dpb    = (hsel == 0) ? 45 : 29;         // att_b, att_w, bn1x4, bn2x4, d1(ob,ow,w), d2(ob,ow,w)

        composition_kernel<<<gP, bP, 0, stream>>>(xp, part0, nparts, F(comp_base + 4),
            F(comp_base + 0), F(comp_base + 1), F(comp_base + 2), F(comp_base + 3), comp_m);
        decomposition_kernel<<<gP, bP, 0, stream>>>(xf, x, F(dec + 5),
            F(dec + 1), F(dec + 2), F(dec + 3), F(dec + 4), F(dec + 6), F(dec + 0),
            dec_m, out_att, hsel);

        // part dependency: dfconv d1 (20->20) -> dfconv d2 (20->10) -> (2-a)*h
        conv3x3_offs_wmma<32><<<gW, bW, 0, stream>>>(T_dp[hsel], P_d1_ow[hsel], F(dpb + 10), offs, zb);
        dfconv_sample_wmma<32, 2, 0><<<gW, bW, 0, stream>>>(T_dp[hsel], offs, P_d1_w[hsel], 20,
            F(dpb + 2), F(dpb + 3), F(dpb + 4), F(dpb + 5), nullptr, T_small, nullptr, zb);
        conv3x3_offs_wmma<32><<<gW, bW, 0, stream>>>(T_small, P_d2_ow[hsel], F(dpb + 13), offs, zb);
        dfconv_sample_wmma<32, 1, 1><<<gW, bW, 0, stream>>>(T_small, offs, P_d2_w[hsel], 10,
            F(dpb + 6), F(dpb + 7), F(dpb + 8), F(dpb + 9), nullptr, nullptr, d2h_m, zb);
        dp_finalize_kernel<<<gP, bP, 0, stream>>>(pA, F(dpb + 1), F(dpb + 0), d2h_m, dp_m);

        // conv_update: cat(fea,x,comp,dec,dp)=296ch -> c1 -> gate/c2 -> combine
        {
            int tot = NTOT * 320;
            build_T_cat5<<<(tot + 255) / 256, 256, 0, stream>>>(fea, x, comp_m, dec_m, dp_m, T_cat);
        }
        conv3x3_offs_wmma<320><<<gW, bW, 0, stream>>>(T_cat, P_c1_ow, F(69), offs, zb);
        dfconv_sample_wmma<320, 2, 0><<<gW, bW, 0, stream>>>(T_cat, offs, P_c1_w, 20,
            F(61), F(62), F(63), F(64), nullptr, T_small, nullptr, zb);
        conv3x3_offs_wmma<32><<<gW, bW, 0, stream>>>(T_small, P_g_ow, F(76), offs, zb);
        dfconv_sample_wmma<32, 1, 2><<<gW, bW, 0, stream>>>(T_small, offs, P_g_w, 1,
            nullptr, nullptr, nullptr, nullptr, F(75), nullptr, gate_m, zb);
        conv3x3_offs_wmma<32><<<gW, bW, 0, stream>>>(T_small, P_c2_ow, F(72), offs, zb);
        dfconv_sample_wmma<32, 1, 1><<<gW, bW, 0, stream>>>(T_small, offs, P_c2_w, 10,
            F(65), F(66), F(67), F(68), nullptr, nullptr, out2_m, zb);
        final_combine_kernel<<<gP, bP, 0, stream>>>(gate_m, x, out2_m,
            (float*)d_out + hsel * 184320);
    }
}
